// NTXentLoss_16673063043753
// MI455X (gfx1250) — matrix operand order repaired and checked by reading, not compiled, not run on hardware
//
#include <hip/hip_runtime.h>

// ---- problem constants (from reference) ----
#define B_N   8192
#define D_K   128
#define TAU_INV     14.285714285714286f          // 1/0.07
#define K1C         20.609929155556620f          // log2(e)/0.07
#define WAVES       4
#define ROWS_PER_WAVE 16
#define ROWS_PER_BLOCK (WAVES * ROWS_PER_WAVE)   // 64
#define NBLOCKS    (B_N / ROWS_PER_BLOCK)        // 128
#define NTILES     (B_N / 16)                    // 512

typedef __attribute__((ext_vector_type(16))) _Float16 v16h;
typedef __attribute__((ext_vector_type(8)))  _Float16 v8h;
typedef __attribute__((ext_vector_type(8)))  float    v8f;

// ---------------------------------------------------------------------------
// Kernel 1: split fp32 embeddings into hi/lo _Float16 planes (error-
// compensated f16: hi+lo carries ~22 mantissa bits ~= fp32).
// ---------------------------------------------------------------------------
__global__ __launch_bounds__(256) void supcon_split_kernel(
    const float* __restrict__ x, _Float16* __restrict__ hi,
    _Float16* __restrict__ lo, int n)
{
  int i = blockIdx.x * 256 + threadIdx.x;
  if (i < n) {
    float f = x[i];
    _Float16 h = (_Float16)f;
    hi[i] = h;
    lo[i] = (_Float16)(f - (float)h);
  }
}

// ---------------------------------------------------------------------------
// Kernel 2: fused  E@E^T (WMMA f32<-f16, hi/lo compensated)  +  per-row
// masked logsumexp / positive-sum / positive-count.
// Each wave owns 16 rows; iterates over all 512 column tiles of 16.
// D-tile layout: acc[r], lanes 0-15 -> (M=r, N=lane), lanes 16-31 -> (M=r+8).
// ---------------------------------------------------------------------------
__global__ __launch_bounds__(WAVES * 32) void supcon_main_kernel(
    const _Float16* __restrict__ ehi, const _Float16* __restrict__ elo,
    const int* __restrict__ labels,
    float* __restrict__ ws_loss, float* __restrict__ ws_has)
{
  __shared__ int slab[B_N];
  for (int idx = threadIdx.x; idx < B_N; idx += WAVES * 32)
    slab[idx] = labels[idx];
  __syncthreads();

  const int lane = threadIdx.x & 31;
  const int wave = threadIdx.x >> 5;
  const int g    = lane >> 4;      // half-wave select
  const int m    = lane & 15;      // row (A) / col (B) within tile
  const int R0   = (blockIdx.x * WAVES + wave) * ROWS_PER_WAVE;

  // ---- preload this wave's A block (16 rows x 128 K), hi and lo planes ----
  // 16-bit A 16x32 layout: lane m holds runs [32c+8g, +8) and [32c+16+8g, +8).
  const _Float16* ahr = ehi + (size_t)(R0 + m) * D_K;
  const _Float16* alr = elo + (size_t)(R0 + m) * D_K;
  v16h a_hi[4], a_lo[4];
#pragma unroll
  for (int c = 0; c < 4; ++c) {
    v8h h0 = *(const v8h*)(ahr + 32 * c + 8 * g);
    v8h h1 = *(const v8h*)(ahr + 32 * c + 16 + 8 * g);
    v8h l0 = *(const v8h*)(alr + 32 * c + 8 * g);
    v8h l1 = *(const v8h*)(alr + 32 * c + 16 + 8 * g);
#pragma unroll
    for (int e = 0; e < 8; ++e) {
      a_hi[c][e] = h0[e]; a_hi[c][e + 8] = h1[e];
      a_lo[c][e] = l0[e]; a_lo[c][e + 8] = l1[e];
    }
  }

  int labr[8];
#pragma unroll
  for (int r = 0; r < 8; ++r) labr[r] = slab[R0 + r + 8 * g];

  float sumexp[8], psum[8], npos[8];
#pragma unroll
  for (int r = 0; r < 8; ++r) { sumexp[r] = 0.f; psum[r] = 0.f; npos[r] = 0.f; }

  for (int jt = 0; jt < NTILES; ++jt) {
    const int C0 = jt * 16;
    // B 32x16 layout: lane n holds runs [32c+8g, +8) and [32c+16+8g, +8) of col C0+n.
    const _Float16* bhr = ehi + (size_t)(C0 + m) * D_K;
    const _Float16* blr = elo + (size_t)(C0 + m) * D_K;

    // Prefetch next tile's B rows (256B/row) -> global_prefetch_b8, hides L2
    // latency under this tile's WMMAs. No counter / register cost.
    if (jt + 1 < NTILES) {
      const char* pnh = (const char*)(bhr + 16 * D_K);
      const char* pnl = (const char*)(blr + 16 * D_K);
      __builtin_prefetch(pnh, 0, 3);
      __builtin_prefetch(pnh + 128, 0, 3);
      __builtin_prefetch(pnl, 0, 3);
      __builtin_prefetch(pnl + 128, 0, 3);
    }

    // Hoist all B loads so they issue as one clause before the WMMA block.
    v16h b_hi[4], b_lo[4];
#pragma unroll
    for (int c = 0; c < 4; ++c) {
      v8h bh0 = *(const v8h*)(bhr + 32 * c + 8 * g);
      v8h bh1 = *(const v8h*)(bhr + 32 * c + 16 + 8 * g);
      v8h bl0 = *(const v8h*)(blr + 32 * c + 8 * g);
      v8h bl1 = *(const v8h*)(blr + 32 * c + 16 + 8 * g);
#pragma unroll
      for (int e = 0; e < 8; ++e) {
        b_hi[c][e] = bh0[e]; b_hi[c][e + 8] = bh1[e];
        b_lo[c][e] = bl0[e]; b_lo[c][e + 8] = bl1[e];
      }
    }

    v8f acc0 = {}, acc1 = {}, acc2 = {};   // 3 independent WMMA chains
#pragma unroll
    for (int c = 0; c < 4; ++c) {
      acc0 = __builtin_amdgcn_wmma_f32_16x16x32_f16(
          false, a_hi[c], false, b_hi[c], (short)0, acc0, false, false);
      acc1 = __builtin_amdgcn_wmma_f32_16x16x32_f16(
          false, a_hi[c], false, b_lo[c], (short)0, acc1, false, false);
      acc2 = __builtin_amdgcn_wmma_f32_16x16x32_f16(
          false, a_lo[c], false, b_hi[c], (short)0, acc2, false, false);
    }

    const int j    = C0 + m;
    const int labc = slab[j];
#pragma unroll
    for (int r = 0; r < 8; ++r) {
      const int i = R0 + r + 8 * g;
      float d = acc0[r] + acc1[r] + acc2[r];          // cosine sim (fp32-ish)
      // exp(d/tau - 1/tau) = exp2(K1*(d-1));  d<=1 since rows normalized,
      // so the fixed shift 1/tau is a valid softmax max -> no online max.
      float t = exp2f(fmaf(d, K1C, -K1C));
      bool diag  = (i == j);
      bool match = (!diag) && (labr[r] == labc);
      sumexp[r] += diag  ? 0.0f : t;
      psum[r]   += match ? d    : 0.0f;
      npos[r]   += match ? 1.0f : 0.0f;
    }
  }

  // ---- reduce 16 column-slots per row (within each half-wave) ----
#pragma unroll
  for (int r = 0; r < 8; ++r) {
#pragma unroll
    for (int s = 1; s < 16; s <<= 1) {
      sumexp[r] += __shfl_xor(sumexp[r], s, 32);
      psum[r]   += __shfl_xor(psum[r],   s, 32);
      npos[r]   += __shfl_xor(npos[r],   s, 32);
    }
  }

  if (m == 0) {   // lanes 0 (rows r) and 16 (rows r+8) write results
#pragma unroll
    for (int r = 0; r < 8; ++r) {
      const int i = R0 + r + 8 * g;
      float np  = npos[r];
      bool  has = np > 0.5f;
      float log_denom = TAU_INV + logf(sumexp[r]);   // fixed max + log(sum)
      float li = -(psum[r] * TAU_INV / fmaxf(np, 1.0f) - log_denom);
      ws_loss[i] = has ? li   : 0.0f;
      ws_has[i]  = has ? 1.0f : 0.0f;
    }
  }
}

// ---------------------------------------------------------------------------
// Kernel 3: deterministic fixed-order final reduction to the scalar loss.
// ---------------------------------------------------------------------------
__global__ __launch_bounds__(256) void supcon_finalize_kernel(
    const float* __restrict__ ws_loss, const float* __restrict__ ws_has,
    float* __restrict__ out)
{
  __shared__ float sl[256], sh[256];
  float l = 0.f, h = 0.f;
  for (int i = threadIdx.x; i < B_N; i += 256) {
    l += ws_loss[i];
    h += ws_has[i];
  }
  sl[threadIdx.x] = l; sh[threadIdx.x] = h;
  __syncthreads();
  for (int s = 128; s > 0; s >>= 1) {
    if ((int)threadIdx.x < s) {
      sl[threadIdx.x] += sl[threadIdx.x + s];
      sh[threadIdx.x] += sh[threadIdx.x + s];
    }
    __syncthreads();
  }
  if (threadIdx.x == 0) out[0] = sl[0] / fmaxf(sh[0], 1.0f);
}

// ---------------------------------------------------------------------------
extern "C" void kernel_launch(void* const* d_in, const int* in_sizes, int n_in,
                              void* d_out, int out_size, void* d_ws, size_t ws_size,
                              hipStream_t stream) {
  const float* emb    = (const float*)d_in[0];   // [8192,128] f32
  const int*   labels = (const int*)d_in[1];     // [8192] int

  // workspace layout: hi plane (2MB) | lo plane (2MB) | loss[8192] | has[8192]
  _Float16* ws_hi  = (_Float16*)d_ws;
  _Float16* ws_lo  = ws_hi + (size_t)B_N * D_K;
  float*    ws_loss = (float*)(ws_lo + (size_t)B_N * D_K);
  float*    ws_has  = ws_loss + B_N;

  const int nelem = B_N * D_K;
  supcon_split_kernel<<<(nelem + 255) / 256, 256, 0, stream>>>(
      emb, ws_hi, ws_lo, nelem);
  supcon_main_kernel<<<NBLOCKS, WAVES * 32, 0, stream>>>(
      ws_hi, ws_lo, labels, ws_loss, ws_has);
  supcon_finalize_kernel<<<1, 256, 0, stream>>>(ws_loss, ws_has, (float*)d_out);
}
